// QuantizableExperts_9783935500531
// MI455X (gfx1250) — matrix operand order repaired and checked by reading, not compiled, not run on hardware
//
#include <hip/hip_runtime.h>
#include <hip/hip_bf16.h>

#define T_TOK 1024
#define H_DIM 1024
#define I_DIM 1408
#define E_NUM 8
#define K_TOP 2
#define NPAIR (T_TOK * K_TOP)

typedef __attribute__((ext_vector_type(16))) __bf16 v16bf;
typedef __attribute__((ext_vector_type(8)))  float  v8f;

union Frag { v16bf v; uint4 q[2]; };

// Packed fp32->bf16 (RNE) via the hardware packed convert. The clang builtin
// is not declared for this toolchain, so emit the instruction directly.
__device__ __forceinline__ unsigned int pkbf(float x, float y) {
#if defined(__gfx1250__)
    unsigned int r;
    asm("v_cvt_pk_bf16_f32 %0, %1, %2" : "=v"(r) : "v"(x), "v"(y));
    return r;
#else
    union { __bf16 b[2]; unsigned int u; } r;
    r.b[0] = (__bf16)x; r.b[1] = (__bf16)y;
    return r.u;
#endif
}
__device__ __forceinline__ unsigned short bf1(float x) {
    return (unsigned short)(pkbf(x, 0.f) & 0xFFFFu);
}
__device__ __forceinline__ uint2 pkbf4(float4 v) {
    uint2 p; p.x = pkbf(v.x, v.y); p.y = pkbf(v.z, v.w); return p;
}

// ---------------------------------------------------------------- zero output
__global__ __launch_bounds__(256) void zero_out_k(float* __restrict__ out) {
    int idx = blockIdx.x * 256 + threadIdx.x;               // 262144 threads * 4 floats = T*H
    float4 z = make_float4(0.f, 0.f, 0.f, 0.f);
    *(float4*)(out + (size_t)idx * 4) = z;
}

// ---------------------------------------------------------------- routing
__global__ __launch_bounds__(256) void route_k(const int* __restrict__ tki,
                                               const float* __restrict__ tkw,
                                               int* __restrict__ counts,
                                               int* __restrict__ offsets,
                                               int* __restrict__ pair_tok,
                                               float* __restrict__ pair_w) {
    __shared__ int s_cnt[E_NUM], s_off[E_NUM], s_pos[E_NUM];
    int tid = threadIdx.x;
    if (tid < E_NUM) s_cnt[tid] = 0;
    __syncthreads();
    for (int t = tid; t < T_TOK; t += 256)
        for (int k = 0; k < K_TOP; ++k)
            atomicAdd(&s_cnt[tki[t * K_TOP + k]], 1);
    __syncthreads();
    if (tid == 0) {
        int acc = 0;
        for (int e = 0; e < E_NUM; ++e) { s_off[e] = acc; s_pos[e] = acc; acc += s_cnt[e]; }
    }
    __syncthreads();
    for (int t = tid; t < T_TOK; t += 256)
        for (int k = 0; k < K_TOP; ++k) {
            int e = tki[t * K_TOP + k];
            int g = atomicAdd(&s_pos[e], 1);
            pair_tok[g] = t;
            pair_w[g]   = tkw[t * K_TOP + k];
        }
    __syncthreads();
    if (tid < E_NUM) { counts[tid] = s_cnt[tid]; offsets[tid] = s_off[tid]; }
}

// ------------------------------------------- grouped GEMM 1: gate_up + silu
// block tile: 32 pair-rows x 64 i-cols; 8 waves -> each wave one 16x16 C tile,
// gate and up computed together (shared A). Software-pipelined: global loads
// for chunk c+1 are issued before the WMMAs of chunk c.
__global__ __launch_bounds__(256) void gateup_silu_k(const float* __restrict__ hs,
                                                     const float* __restrict__ gup,
                                                     const int* __restrict__ counts,
                                                     const int* __restrict__ offsets,
                                                     const int* __restrict__ pair_tok,
                                                     unsigned short* __restrict__ hbuf) {
    const int e   = blockIdx.z;
    const int cnt = counts[e];
    const int off = offsets[e];
    const int mtile = blockIdx.y;
    if (mtile * 32 >= cnt) return;
    const int j0 = blockIdx.x * 64;

    __shared__ int            s_tok[32];
    __shared__ unsigned short As[32][32];
    __shared__ unsigned short Bg[64][32];
    __shared__ unsigned short Bu[64][32];

    const int tid  = threadIdx.x;
    const int wave = tid >> 5;
    const int lane = tid & 31;
    const int half = lane >> 4;
    const int mn   = lane & 15;
    const int wm   = wave & 1;      // M sub-tile (0..1)
    const int wn   = wave >> 1;     // N sub-tile (0..3)

    if (tid < 32) {
        int m = mtile * 32 + tid;
        s_tok[tid] = (m < cnt) ? pair_tok[off + m] : -1;
    }
    __syncthreads();

    const float* Wg = gup + (size_t)e * (2 * I_DIM) * H_DIM;
    const int row = tid >> 3, kp = tid & 7;     // staging coordinates
    const int tok = s_tok[row];
    const float* pa  = (tok >= 0) ? &hs[(size_t)tok * H_DIM + kp * 4] : nullptr;
    const float* pg0 = &Wg[(size_t)(j0 + row) * H_DIM + kp * 4];
    const float* pg1 = &Wg[(size_t)(j0 + 32 + row) * H_DIM + kp * 4];
    const float* pu0 = &Wg[(size_t)(I_DIM + j0 + row) * H_DIM + kp * 4];
    const float* pu1 = &Wg[(size_t)(I_DIM + j0 + 32 + row) * H_DIM + kp * 4];

    float4 ra, rg0, rg1, ru0, ru1;
    auto load_stage = [&](int k0) {
        ra = make_float4(0.f, 0.f, 0.f, 0.f);
        if (pa) ra = *(const float4*)(pa + k0);
        rg0 = *(const float4*)(pg0 + k0);
        rg1 = *(const float4*)(pg1 + k0);
        ru0 = *(const float4*)(pu0 + k0);
        ru1 = *(const float4*)(pu1 + k0);
        if (k0 + 32 < H_DIM) {                  // prefetch one further chunk ahead
            __builtin_prefetch(pg0 + k0 + 32, 0, 1);
            __builtin_prefetch(pu0 + k0 + 32, 0, 1);
        }
    };
    auto store_stage = [&]() {
        *(uint2*)&As[row][kp * 4]      = pkbf4(ra);
        *(uint2*)&Bg[row][kp * 4]      = pkbf4(rg0);
        *(uint2*)&Bg[32 + row][kp * 4] = pkbf4(rg1);
        *(uint2*)&Bu[row][kp * 4]      = pkbf4(ru0);
        *(uint2*)&Bu[32 + row][kp * 4] = pkbf4(ru1);
    };

    v8f accg = {}; v8f accu = {};
    const int arow = wm * 16 + mn;              // lane's A row
    const int bn   = wn * 16 + mn;              // lane's B column
    const int kb   = half * 16;                 // lane's B K-range

    load_stage(0);
    for (int c = 0; c < H_DIM / 32; ++c) {
        store_stage();
        __syncthreads();
        if (c + 1 < H_DIM / 32) load_stage((c + 1) * 32);   // overlap with WMMAs below

        Frag a, bg, bu;
        a.q[0]  = *(const uint4*)&As[arow][half * 16];       // K = half*16 .. +7
        a.q[1]  = *(const uint4*)&As[arow][half * 16 + 8];   // K = half*16+8 .. +7
        bg.q[0] = *(const uint4*)&Bg[bn][kb];
        bg.q[1] = *(const uint4*)&Bg[bn][kb + 8];
        bu.q[0] = *(const uint4*)&Bu[bn][kb];
        bu.q[1] = *(const uint4*)&Bu[bn][kb + 8];

        accg = __builtin_amdgcn_wmma_f32_16x16x32_bf16(false, a.v, false, bg.v,
                                                       (short)0, accg, false, false);
        accu = __builtin_amdgcn_wmma_f32_16x16x32_bf16(false, a.v, false, bu.v,
                                                       (short)0, accu, false, false);
        __syncthreads();
    }

    // ---- epilogue: h = silu(gate) * up -> bf16 hbuf
    for (int r = 0; r < 8; ++r) {
        int m  = wm * 16 + r + 8 * half;         // local row 0..31
        int gm = mtile * 32 + m;
        if (gm >= cnt) continue;
        float g = accg[r], u = accu[r];
        float h = g / (1.f + __expf(-g)) * u;
        int col = j0 + wn * 16 + mn;
        hbuf[(size_t)(off + gm) * I_DIM + col] = bf1(h);
    }
}

// ------------------------------------------- grouped GEMM 2: down proj + combine
__global__ __launch_bounds__(256) void down_comb_k(const unsigned short* __restrict__ hbuf,
                                                   const float* __restrict__ dwn,
                                                   const int* __restrict__ counts,
                                                   const int* __restrict__ offsets,
                                                   const int* __restrict__ pair_tok,
                                                   const float* __restrict__ pair_w,
                                                   float* __restrict__ out) {
    const int e   = blockIdx.z;
    const int cnt = counts[e];
    const int off = offsets[e];
    const int mtile = blockIdx.y;
    if (mtile * 32 >= cnt) return;
    const int n0 = blockIdx.x * 64;

    __shared__ unsigned short As[32][32];
    __shared__ unsigned short Bs[64][32];

    const int tid  = threadIdx.x;
    const int wave = tid >> 5;
    const int lane = tid & 31;
    const int half = lane >> 4;
    const int mn   = lane & 15;
    const int wm   = wave & 1;
    const int wn   = wave >> 1;

    const float* Wd = dwn + (size_t)e * H_DIM * I_DIM;
    const int row = tid >> 3, kp = tid & 7;
    const int m = mtile * 32 + row;
    const unsigned short* pa = (m < cnt) ? &hbuf[(size_t)(off + m) * I_DIM + kp * 4] : nullptr;
    const float* pb0 = &Wd[(size_t)(n0 + row) * I_DIM + kp * 4];
    const float* pb1 = &Wd[(size_t)(n0 + 32 + row) * I_DIM + kp * 4];

    uint2 rha; float4 rb0, rb1;
    auto load_stage = [&](int k0) {
        rha = make_uint2(0u, 0u);
        if (pa) rha = *(const uint2*)(pa + k0);
        rb0 = *(const float4*)(pb0 + k0);
        rb1 = *(const float4*)(pb1 + k0);
        if (k0 + 32 < I_DIM) __builtin_prefetch(pb0 + k0 + 32, 0, 1);
    };
    auto store_stage = [&]() {
        *(uint2*)&As[row][kp * 4]      = rha;
        *(uint2*)&Bs[row][kp * 4]      = pkbf4(rb0);
        *(uint2*)&Bs[32 + row][kp * 4] = pkbf4(rb1);
    };

    v8f acc = {};
    const int arow = wm * 16 + mn;
    const int bn   = wn * 16 + mn;
    const int kb   = half * 16;

    load_stage(0);
    for (int c = 0; c < I_DIM / 32; ++c) {
        store_stage();
        __syncthreads();
        if (c + 1 < I_DIM / 32) load_stage((c + 1) * 32);

        Frag a, b;
        a.q[0] = *(const uint4*)&As[arow][half * 16];
        a.q[1] = *(const uint4*)&As[arow][half * 16 + 8];
        b.q[0] = *(const uint4*)&Bs[bn][kb];
        b.q[1] = *(const uint4*)&Bs[bn][kb + 8];

        acc = __builtin_amdgcn_wmma_f32_16x16x32_bf16(false, a.v, false, b.v,
                                                      (short)0, acc, false, false);
        __syncthreads();
    }

    // ---- epilogue: scale by routing weight, combine into out (exactly K=2
    // commutative adds onto 0.0 per element -> deterministic)
    for (int r = 0; r < 8; ++r) {
        int mm = wm * 16 + r + 8 * half;
        int gm = mtile * 32 + mm;
        if (gm >= cnt) continue;
        float w = pair_w[off + gm];
        int   t = pair_tok[off + gm];
        int col = n0 + wn * 16 + mn;
        atomicAdd(&out[(size_t)t * H_DIM + col], acc[r] * w);
    }
}

extern "C" void kernel_launch(void* const* d_in, const int* in_sizes, int n_in,
                              void* d_out, int out_size, void* d_ws, size_t ws_size,
                              hipStream_t stream) {
    (void)in_sizes; (void)n_in; (void)out_size; (void)ws_size;
    const float* hs  = (const float*)d_in[0];   // hidden_states (T,H) f32
    const float* tkw = (const float*)d_in[1];   // top_k_weights (T,K) f32
    const float* gup = (const float*)d_in[2];   // gate_up_proj (E,2I,H) f32
    const float* dwn = (const float*)d_in[3];   // down_proj (E,H,I) f32
    const int*   tki = (const int*)d_in[4];     // top_k_index (T,K)
    float* out = (float*)d_out;

    char* ws = (char*)d_ws;
    int*   counts   = (int*)(ws);
    int*   offsets  = (int*)(ws + 64);
    int*   pair_tok = (int*)(ws + 128);
    float* pair_w   = (float*)(ws + 128 + NPAIR * 4);
    unsigned short* hbuf = (unsigned short*)(ws + 128 + 2 * NPAIR * 4 + 64); // 16B aligned

    zero_out_k<<<dim3((T_TOK * H_DIM) / (256 * 4)), 256, 0, stream>>>(out);
    route_k<<<1, 256, 0, stream>>>(tki, tkw, counts, offsets, pair_tok, pair_w);

    dim3 g2(I_DIM / 64, NPAIR / 32, E_NUM);     // 22 x 64 x 8, early-exit on empty M tiles
    gateup_silu_k<<<g2, 256, 0, stream>>>(hs, gup, counts, offsets, pair_tok, hbuf);

    dim3 g3(H_DIM / 64, NPAIR / 32, E_NUM);     // 16 x 64 x 8
    down_comb_k<<<g3, 256, 0, stream>>>(hbuf, dwn, counts, offsets, pair_tok, pair_w, out);
}
